// MultiHeadAttention_80607946211361
// MI455X (gfx1250) — compile-verified, matrix operand order checked by reading, not run on hardware
//
#include <hip/hip_runtime.h>
#include <cstdint>
#include <cstddef>

// ---------------------------------------------------------------------------
// MultiHeadAttention forward for MI455X (gfx1250, wave32, WMMA).
// B=2, S=2048, D=1024, H=16, depth=64.
// All matmuls via v_wmma_f32_16x16x32_f16 (f16 inputs, f32 accumulate).
// The 537MB weights tensor is written to HBM exactly once (bandwidth floor);
// softmax stats come from a recompute pass (online max/sum, no logits buffer).
// V is kept transposed [b,h,64,S] so P@V B-fragments are contiguous b128 loads.
// ---------------------------------------------------------------------------

typedef _Float16 f16;
typedef __attribute__((ext_vector_type(16))) _Float16 v16h;
typedef __attribute__((ext_vector_type(8)))  float    v8f;

#define WMMA_FF16(a, b, c) \
  __builtin_amdgcn_wmma_f32_16x16x32_f16(false, (a), false, (b), (short)0, (c), false, false)

static constexpr int Bn    = 2;
static constexpr int Sn    = 2048;
static constexpr int Dn    = 1024;
static constexpr int Hn    = 16;
static constexpr int DPn   = 64;
static constexpr int Mrows = Bn * Sn;        // 4096
static constexpr float QK_SCALE = 0.125f;    // 1/sqrt(64)

// A/B fragment loader (16x32 f16) from row-major [16][>=32] with row stride ld.
// lane l: m = l&15; halves 0..7 -> K=(l>>4)*8+i, halves 8..15 -> K=16+(l>>4)*8+i
__device__ inline v16h frag_ld_rows(const f16* base, int ld) {
  const int l  = threadIdx.x & 31;
  const int m  = l & 15;
  const int kb = (l >> 4) * 8;
  v16h f;
#pragma unroll
  for (int i = 0; i < 8; ++i) {
    f[i]     = base[m * ld + kb + i];
    f[8 + i] = base[m * ld + 16 + kb + i];
  }
  return f;
}

// -------------------- fp32 -> f16 weight conversion (x4 vectorized) --------
__global__ __launch_bounds__(256) void wcvt_kernel(
    const float* __restrict__ wq, const float* __restrict__ wk,
    const float* __restrict__ wv, const float* __restrict__ wo,
    f16* __restrict__ dst) {
  int i4 = blockIdx.x * 256 + threadIdx.x;     // 0 .. 4*1048576/4 - 1
  int i  = i4 * 4;
  int which = i >> 20;
  int off   = i & 1048575;
  const float* src = (which == 0) ? wq : (which == 1) ? wk
                   : (which == 2) ? wv : wo;
  float4 t = *(const float4*)(src + off);
  alignas(8) f16 h[4] = { (f16)t.x, (f16)t.y, (f16)t.z, (f16)t.w };
  *(uint2*)(dst + i) = *(const uint2*)h;
}

// -------------------- tiled GEMM: out = X @ W^T + bias ---------------------
// X: fp32 [4096][1024]; W: f16 [1024][1024] row-major (rows of B^T).
// Block tile 128x128, 8 waves (4x2), wave tile 32x64 (2x4 WMMA tiles),
// K step 32, double-buffered LDS with fetch->compute->stage schedule.
// STORE_MODE 0: f16 head-split [B,H,S,64]      (Q/K projections)
// STORE_MODE 1: fp32 flat [4096][1024]         (output projection)
// STORE_MODE 2: f16 transposed [B,H,64,S]      (V projection, b128 stores)
template <int STORE_MODE>
__global__ __launch_bounds__(256) void gemm_kernel(
    const float* __restrict__ X, const f16* __restrict__ W,
    const float* __restrict__ bias,
    f16* __restrict__ outH, float* __restrict__ outF) {
  __shared__ f16 As[2][128 * 32];   // 2 x 8 KB
  __shared__ f16 Bs[2][128 * 32];   // 2 x 8 KB

  const int tid  = threadIdx.x;
  const int lane = tid & 31;
  const int wid  = tid >> 5;
  const int wm   = wid & 3;                  // wave row (32 rows each)
  const int wn   = wid >> 2;                 // wave col (64 cols each)
  const int m0   = blockIdx.y * 128;
  const int n0   = blockIdx.x * 128;
  const int r    = tid >> 1;                 // staging row 0..127
  const int c0   = (tid & 1) * 16;           // staging col 0 or 16

  v8f acc[2][4] = {};

  // Prologue: stage K-tile 0 into buffer 0.
  {
    const float* xs = X + (size_t)(m0 + r) * Dn + c0;
    f16* ad = As[0] + r * 32 + c0;
#pragma unroll
    for (int i = 0; i < 16; i += 4) {
      float4 t = *(const float4*)(xs + i);
      ad[i + 0] = (f16)t.x; ad[i + 1] = (f16)t.y;
      ad[i + 2] = (f16)t.z; ad[i + 3] = (f16)t.w;
    }
    const uint4* wsrc = (const uint4*)(W + (size_t)(n0 + r) * Dn + c0);
    uint4* wd = (uint4*)(Bs[0] + r * 32 + c0);
    wd[0] = wsrc[0]; wd[1] = wsrc[1];
  }
  __syncthreads();

  for (int kk = 0; kk < Dn / 32; ++kk) {
    const int  cur  = kk & 1;
    const bool more = (kk + 1) < (Dn / 32);
    // 1) fetch next K-tile into registers (latency overlaps the WMMAs below)
    float4 xr[4];
    uint4  wr[2];
    if (more) {
      const int k0 = (kk + 1) * 32;
      const float* xs = X + (size_t)(m0 + r) * Dn + k0 + c0;
#pragma unroll
      for (int i = 0; i < 4; ++i) xr[i] = *(const float4*)(xs + i * 4);
      const uint4* wsrc = (const uint4*)(W + (size_t)(n0 + r) * Dn + k0 + c0);
      wr[0] = wsrc[0]; wr[1] = wsrc[1];
    }
    // 2) compute on current buffer: 8 WMMAs
    v16h a0 = frag_ld_rows(As[cur] + (wm * 32 + 0)  * 32, 32);
    v16h a1 = frag_ld_rows(As[cur] + (wm * 32 + 16) * 32, 32);
#pragma unroll
    for (int j = 0; j < 4; ++j) {
      v16h bf = frag_ld_rows(Bs[cur] + (wn * 64 + j * 16) * 32, 32);
      acc[0][j] = WMMA_FF16(a0, bf, acc[0][j]);
      acc[1][j] = WMMA_FF16(a1, bf, acc[1][j]);
    }
    // 3) stage fetched tile into the other buffer
    if (more) {
      f16* ad = As[cur ^ 1] + r * 32 + c0;
#pragma unroll
      for (int i = 0; i < 4; ++i) {
        ad[i * 4 + 0] = (f16)xr[i].x; ad[i * 4 + 1] = (f16)xr[i].y;
        ad[i * 4 + 2] = (f16)xr[i].z; ad[i * 4 + 3] = (f16)xr[i].w;
      }
      uint4* wd = (uint4*)(Bs[cur ^ 1] + r * 32 + c0);
      wd[0] = wr[0]; wd[1] = wr[1];
    }
    __syncthreads();
  }

  // Epilogue (branchless per instantiation).
  // C layout: lane holds column (lane&15); VGPR rr -> row rr + 8*(lane>>4).
  const int col   = lane & 15;
  const int rbase = (lane >> 4) * 8;
#pragma unroll
  for (int i = 0; i < 2; ++i) {
#pragma unroll
    for (int j = 0; j < 4; ++j) {
      const int   nt    = n0 + wn * 64 + j * 16 + col;
      const float bv    = bias[nt];
      const int   mbase = m0 + wm * 32 + i * 16 + rbase;
      if (STORE_MODE == 1) {
#pragma unroll
        for (int rr = 0; rr < 8; ++rr)
          outF[(size_t)(mbase + rr) * Dn + nt] = acc[i][j][rr] + bv;
      } else if (STORE_MODE == 0) {
#pragma unroll
        for (int rr = 0; rr < 8; ++rr) {
          const int mt = mbase + rr;
          const int bb = mt >> 11, ss = mt & 2047;
          const int hh = nt >> 6,  dd = nt & 63;
          outH[(((size_t)(bb * Hn + hh)) * Sn + ss) * DPn + dd] =
              (f16)(acc[i][j][rr] + bv);
        }
      } else {  // STORE_MODE == 2: V transposed [b,h,64,S], one b128 per lane
        alignas(16) f16 hbuf[8];
#pragma unroll
        for (int rr = 0; rr < 8; ++rr) hbuf[rr] = (f16)(acc[i][j][rr] + bv);
        const int bb = mbase >> 11, ss = mbase & 2047;   // 8-aligned ss
        const int hh = nt >> 6,     dd = nt & 63;
        *(uint4*)(outH + (((size_t)(bb * Hn + hh)) * DPn + dd) * Sn + ss) =
            *(const uint4*)hbuf;
      }
    }
  }
}

// -------------------- attention: softmax(QK^T/8 + mask) @ V ----------------
// One wave per 16-query tile. Pass 1: WMMA logits -> online (max, sum).
// Pass 2: recompute logits, write normalized weights once, accumulate
// attn = P @ V (P transposed C->A layout via a 1KB per-wave LDS tile;
// V fragments load contiguously from the transposed Vt tensor).
__global__ __launch_bounds__(128) void attn_kernel(
    const f16* __restrict__ Qp, const f16* __restrict__ Kp,
    const f16* __restrict__ Vt, const float* __restrict__ mask,
    float* __restrict__ Wout, float* __restrict__ Cout) {
  __shared__ f16 Pls[4][16 * 32];   // per-wave probability tile (A-layout src)

  const int lane  = threadIdx.x & 31;
  const int w     = threadIdx.x >> 5;
  const int gt    = blockIdx.x * 4 + w;   // 0..4095 query tiles
  const int bh    = gt >> 7;              // b*H + h   (S/16 = 128 tiles each)
  const int qt    = gt & 127;
  const int b     = bh >> 4;
  const int h     = bh & 15;
  const int col   = lane & 15;
  const int rbase = (lane >> 4) * 8;

  const f16* qbase  = Qp + ((size_t)bh * Sn + qt * 16) * DPn;
  const f16* vtbase = Vt + (size_t)bh * DPn * Sn;
  const v16h aq0 = frag_ld_rows(qbase, DPn);        // depth 0..31
  const v16h aq1 = frag_ld_rows(qbase + 32, DPn);   // depth 32..63

  float mrow[8], srow[8];
#pragma unroll
  for (int rr = 0; rr < 8; ++rr) { mrow[rr] = -3.0e38f; srow[rr] = 0.0f; }

  // ---- Pass 1: per-row online max + rescaled sum ----
  for (int kt = 0; kt < Sn / 32; ++kt) {
    __builtin_prefetch(Kp + ((size_t)bh * Sn + kt * 32 + 128) * DPn, 0, 1);
#pragma unroll
    for (int t = 0; t < 2; ++t) {
      const int kidx0 = kt * 32 + t * 16;
      const f16* kb = Kp + ((size_t)bh * Sn + kidx0) * DPn;
      v8f c = {};
      c = WMMA_FF16(aq0, frag_ld_rows(kb, DPn), c);
      c = WMMA_FF16(aq1, frag_ld_rows(kb + 32, DPn), c);
      const float mk = mask[b * Sn + kidx0 + col] * -1e9f;
#pragma unroll
      for (int rr = 0; rr < 8; ++rr) {
        float x  = c[rr] * QK_SCALE + mk;
        float nm = fmaxf(mrow[rr], x);
        srow[rr] = srow[rr] * __expf(mrow[rr] - nm) + __expf(x - nm);
        mrow[rr] = nm;
      }
    }
  }
  // Cross-lane combine over 16-lane halves (xor 1,2,4,8 stays in-group).
#pragma unroll
  for (int rr = 0; rr < 8; ++rr) {
    float m = mrow[rr], s = srow[rr];
#pragma unroll
    for (int off = 1; off < 16; off <<= 1) {
      float mo = __shfl_xor(m, off, 32);
      float so = __shfl_xor(s, off, 32);
      float nm = fmaxf(m, mo);
      s = s * __expf(m - nm) + so * __expf(mo - nm);
      m = nm;
    }
    mrow[rr] = m;
    srow[rr] = 1.0f / s;   // keep inverse sum
  }

  // ---- Pass 2: write weights once, accumulate attn = P @ V ----
  v8f cav[4] = {};
  for (int kt = 0; kt < Sn / 32; ++kt) {
    __builtin_prefetch(Kp + ((size_t)bh * Sn + kt * 32 + 128) * DPn, 0, 1);
#pragma unroll
    for (int t = 0; t < 2; ++t) {
      const int kidx0 = kt * 32 + t * 16;
      const f16* kb = Kp + ((size_t)bh * Sn + kidx0) * DPn;
      v8f c = {};
      c = WMMA_FF16(aq0, frag_ld_rows(kb, DPn), c);
      c = WMMA_FF16(aq1, frag_ld_rows(kb + 32, DPn), c);
      const float mk  = mask[b * Sn + kidx0 + col] * -1e9f;
      const int   key = kidx0 + col;
#pragma unroll
      for (int rr = 0; rr < 8; ++rr) {
        float x = c[rr] * QK_SCALE + mk;
        float p = __expf(x - mrow[rr]) * srow[rr];   // normalized prob
        const int row = rr + rbase;
        Wout[((size_t)(bh * Sn + qt * 16 + row)) * Sn + key] = p;
        Pls[w][row * 32 + t * 16 + col] = (f16)p;
      }
    }
    __syncthreads();   // Pls writes -> fragment reads (uniform trip count)
    const v16h pa = frag_ld_rows(Pls[w], 32);        // P tile 16x32, A layout
#pragma unroll
    for (int dc = 0; dc < 4; ++dc) {                 // Vt rows are contiguous
      const v16h bv = frag_ld_rows(vtbase + (size_t)(dc * 16) * Sn + kt * 32, Sn);
      cav[dc] = WMMA_FF16(pa, bv, cav[dc]);
    }
    __syncthreads();   // reads done before next iteration overwrites Pls
  }

  // Store attn into concat layout [B, S, H*64] (fp32).
#pragma unroll
  for (int dc = 0; dc < 4; ++dc) {
#pragma unroll
    for (int rr = 0; rr < 8; ++rr) {
      const int srow_i = qt * 16 + rr + rbase;
      Cout[((size_t)(b * Sn + srow_i)) * Dn + h * DPn + dc * 16 + col] =
          cav[dc][rr];
    }
  }
}

// ---------------------------------------------------------------------------
extern "C" void kernel_launch(void* const* d_in, const int* in_sizes, int n_in,
                              void* d_out, int out_size, void* d_ws, size_t ws_size,
                              hipStream_t stream) {
  const float* q    = (const float*)d_in[0];
  const float* k    = (const float*)d_in[1];
  const float* v    = (const float*)d_in[2];
  const float* mask = (const float*)d_in[3];
  const float* wq_w = (const float*)d_in[4];
  const float* wq_b = (const float*)d_in[5];
  const float* wk_w = (const float*)d_in[6];
  const float* wk_b = (const float*)d_in[7];
  const float* wv_w = (const float*)d_in[8];
  const float* wv_b = (const float*)d_in[9];
  const float* wo_w = (const float*)d_in[10];
  const float* wo_b = (const float*)d_in[11];

  float* out     = (float*)d_out;                         // [B,S,D]
  float* weights = (float*)d_out + (size_t)Bn * Sn * Dn;  // [B,H,S,S]

  // Workspace carve-up (~48 MB):
  //  [0]      4 x 1M halves: f16 wq/wk/wv/wo                      (8 MB)
  //  [+8MB]   Qp,Kp f16 [B,H,S,64]; Vt f16 [B,H,64,S]             (3 x 8 MB)
  //  [+32MB]  concat fp32 [B,S,D]                                 (16 MB)
  f16* wAll = (f16*)d_ws;
  f16* Qp = wAll + (size_t)4 * 1048576;
  f16* Kp = Qp + (size_t)Bn * Hn * Sn * DPn;
  f16* Vt = Kp + (size_t)Bn * Hn * Sn * DPn;
  float* concat = (float*)(Vt + (size_t)Bn * Hn * Sn * DPn);

  wcvt_kernel<<<(4 * 1048576 / 4) / 256, 256, 0, stream>>>(wq_w, wk_w, wv_w,
                                                           wo_w, wAll);

  dim3 gg(Dn / 128, Mrows / 128);   // (8, 32)
  gemm_kernel<0><<<gg, 256, 0, stream>>>(q, wAll + 0 * 1048576, wq_b, Qp, nullptr);
  gemm_kernel<0><<<gg, 256, 0, stream>>>(k, wAll + 1 * 1048576, wk_b, Kp, nullptr);
  gemm_kernel<2><<<gg, 256, 0, stream>>>(v, wAll + 2 * 1048576, wv_b, Vt, nullptr);

  attn_kernel<<<(Bn * Hn * (Sn / 16)) / 4, 128, 0, stream>>>(Qp, Kp, Vt, mask,
                                                             weights, concat);

  gemm_kernel<1><<<gg, 256, 0, stream>>>(concat, wAll + 3 * 1048576, wo_b,
                                         nullptr, out);
}
